// HEGNNModel_70265664962767
// MI455X (gfx1250) — compile-verified, matrix operand order checked by reading, not run on hardware
//
#include <hip/hip_runtime.h>
#include <hip/hip_bf16.h>

// ---------------------------------------------------------------------------
// HEGNN forward for MI455X (gfx1250, wave32, WMMA).
// All >=128-wide GEMMs run on v_wmma_f32_16x16x32_bf16 (bf16 in, f32 accum),
// with the weight tile staged in LDS (shared by all 8 waves of the block).
// ---------------------------------------------------------------------------

typedef __bf16 bf16;
typedef __attribute__((ext_vector_type(16))) __bf16 v16bf;
typedef __attribute__((ext_vector_type(8)))  __bf16 v8bf;
typedef __attribute__((ext_vector_type(8)))  float  v8f;

#define N_NODES 10000
#define N_EDGES 160000
#define N_GRAPH 100
#define DIM     128
#define NLAYERS 5
#define EC      32000      // edge chunk (multiple of 128); 5 chunks cover 160000
#define NCHUNKS 5
#define NPAD    10112      // N_NODES rounded up to multiple of 128
#define K1PAD   288        // 257/260 padded to multiple of 32
#define MAXK    288        // max K of any GEMM (LDS weight tile size)
#define SQ3     1.7320508075688772f

__device__ __forceinline__ float silu_f(float x) { return x / (1.f + __expf(-x)); }

__device__ __forceinline__ void sph9(float x, float y, float z, float* o) {
  float r = sqrtf(x * x + y * y + z * z);
  float inv = 1.f / fmaxf(r, 1e-12f);
  x *= inv; y *= inv; z *= inv;
  o[0] = 1.f; o[1] = x; o[2] = y; o[3] = z;
  o[4] = SQ3 * x * z;
  o[5] = SQ3 * x * y;
  o[6] = y * y - 0.5f * (x * x + z * z);
  o[7] = SQ3 * y * z;
  o[8] = 0.5f * SQ3 * (z * z - x * x);
}

// ---------------------------------------------------------------------------
__global__ void k_zero(float* p, long n) {
  long i = (long)blockIdx.x * 256 + threadIdx.x;
  if (i < n) p[i] = 0.f;
}

// W (K x 128 row-major, f32) -> Wt (128 x Kpad, bf16), zero-padded K tail.
__global__ void k_pack_wt(const float* __restrict__ W, bf16* __restrict__ Wt,
                          int K, int Kpad) {
  int i = blockIdx.x * 256 + threadIdx.x;
  if (i >= DIM * Kpad) return;
  int n = i / Kpad, k = i % Kpad;
  Wt[i] = (k < K) ? (bf16)W[(size_t)k * DIM + n] : (bf16)0.f;
}

__global__ void k_embed(const int* __restrict__ atoms, const int* __restrict__ batch,
                        const float* __restrict__ emb, const float* __restrict__ pos,
                        float* __restrict__ feat, bf16* __restrict__ featb,
                        float* __restrict__ pos_cur, float* __restrict__ gsum,
                        float* __restrict__ gcnt) {
  int i = blockIdx.x * 256 + threadIdx.x;
  if (i < N_NODES * DIM) {
    int n = i >> 7, j = i & 127;
    float v = emb[(size_t)atoms[n] * DIM + j];
    feat[i] = v;
    featb[i] = (bf16)v;
  }
  if (i < N_NODES) {
    float px = pos[i * 3], py = pos[i * 3 + 1], pz = pos[i * 3 + 2];
    pos_cur[i * 3] = px; pos_cur[i * 3 + 1] = py; pos_cur[i * 3 + 2] = pz;
    int g = batch[i];
    atomicAdd(&gsum[g * 3 + 0], px);
    atomicAdd(&gsum[g * 3 + 1], py);
    atomicAdd(&gsum[g * 3 + 2], pz);
    atomicAdd(&gcnt[g], 1.f);
  }
}

__global__ void k_edge_count(const int* __restrict__ row, float* __restrict__ cnt) {
  int e = blockIdx.x * 256 + threadIdx.x;
  if (e < N_EDGES) atomicAdd(&cnt[row[e]], 1.f);
}

// X0 row = [dist | feat[row] | feat[col] | pad] (K1PAD bf16)
__global__ void k_build_x0(const int* __restrict__ row, const int* __restrict__ col,
                           const float* __restrict__ pos, const bf16* __restrict__ featb,
                           bf16* __restrict__ X, int ebase, int ecount) {
  long i = (long)blockIdx.x * 256 + threadIdx.x;
  if (i >= (long)ecount * K1PAD) return;
  int el = (int)(i / K1PAD), j = (int)(i % K1PAD);
  int e = ebase + el;
  int r = row[e], c = col[e];
  bf16 v = (bf16)0.f;
  if (j == 0) {
    float dx = pos[r * 3] - pos[c * 3];
    float dy = pos[r * 3 + 1] - pos[c * 3 + 1];
    float dz = pos[r * 3 + 2] - pos[c * 3 + 2];
    v = (bf16)sqrtf(dx * dx + dy * dy + dz * dz);
  } else if (j <= DIM) {
    v = featb[(size_t)r * DIM + (j - 1)];
  } else if (j <= 2 * DIM) {
    v = featb[(size_t)c * DIM + (j - 1 - DIM)];
  }
  X[i] = v;
}

// ---------------------------------------------------------------------------
// WMMA GEMM: C[M x 128] = act(A[M x K] @ W[K x 128] + bias)
// A row-major bf16 (lda elems), W pre-packed transposed bf16: Wt[n][K],
// staged into LDS once per block. Fragment layout per CDNA5 ISA:
//   lane<16 : rows m0+lane,   K = {k0..k0+7, k0+16..k0+23}
//   lane>=16: rows m0+lane-16,K = {k0+8..k0+15, k0+24..k0+31}
// Each wave computes a 16x128 strip (8 accumulators); block = 8 waves = 128 rows.
// B fragments are consumed in two groups of 4 (one live v16bf[4] array) so the
// peak register set (~104 VGPRs) never spills; sched_barrier(0) keeps each
// load group ahead of its WMMA group, and the second load group interleaves
// with the first WMMA group (WAR-only deps) to overlap DS latency with XDL.
// ---------------------------------------------------------------------------
__device__ __forceinline__ v16bf load_frag16x32(const bf16* p) {
  v8bf lo = *(const v8bf*)p;
  v8bf hi = *(const v8bf*)(p + 16);
  v16bf f;
#pragma unroll
  for (int i = 0; i < 8; ++i) { f[i] = lo[i]; f[i + 8] = hi[i]; }
  return f;
}

__global__ void k_gemm_n128(const bf16* __restrict__ A, int lda, int M,
                            const bf16* __restrict__ Wt, int K,
                            const float* __restrict__ bias,
                            bf16* __restrict__ Obf, float* __restrict__ Of, int act) {
  __shared__ __align__(16) bf16 ldsW[DIM * MAXK];  // 73.7 KB of the 320 KB WGP LDS

  // cooperative copy of the weight tile (128 x K bf16) into LDS
  {
    int nv = DIM * K / 8;  // number of 16B vectors
    const v8bf* src = (const v8bf*)Wt;
    v8bf* dst = (v8bf*)ldsW;
    for (int t = threadIdx.x; t < nv; t += 256) dst[t] = src[t];
  }
  __syncthreads();

  int lane = threadIdx.x & 31;
  int wave = threadIdx.x >> 5;
  int m0 = blockIdx.x * 128 + wave * 16;
  int nlo = lane & 15;
  int rA = m0 + nlo;
  int kb = (lane >> 4) << 3;  // 0 or 8

  v8f acc[8];
#pragma unroll
  for (int nf = 0; nf < 8; ++nf)
#pragma unroll
    for (int v = 0; v < 8; ++v) acc[nf][v] = 0.f;

  const bf16* aRow = A + (size_t)rA * lda + kb;
  const bf16* wRow = ldsW + (size_t)nlo * K + kb;

#pragma clang loop unroll(disable)
  for (int k0 = 0; k0 < K; k0 += 32) {
    v16bf a = load_frag16x32(aRow + k0);
    v16bf b[4];
#pragma unroll
    for (int nf = 0; nf < 4; ++nf)
      b[nf] = load_frag16x32(wRow + (size_t)nf * 16 * K + k0);
    __builtin_amdgcn_sched_barrier(0);
#pragma unroll
    for (int nf = 0; nf < 4; ++nf)
      acc[nf] = __builtin_amdgcn_wmma_f32_16x16x32_bf16(
          false, a, false, b[nf], (short)0, acc[nf], false, false);
#pragma unroll
    for (int nf = 0; nf < 4; ++nf)
      b[nf] = load_frag16x32(wRow + (size_t)(nf + 4) * 16 * K + k0);
    __builtin_amdgcn_sched_barrier(0);
#pragma unroll
    for (int nf = 0; nf < 4; ++nf)
      acc[nf + 4] = __builtin_amdgcn_wmma_f32_16x16x32_bf16(
          false, a, false, b[nf], (short)0, acc[nf + 4], false, false);
  }

  // D layout: lane<16 -> N=lane, M=v ; lane>=16 -> N=lane-16, M=v+8
  int mo = (lane >> 4) << 3;
#pragma unroll
  for (int nf = 0; nf < 8; ++nf) {
    int n = nf * 16 + nlo;
    float bv = bias ? bias[n] : 0.f;
#pragma unroll
    for (int v = 0; v < 8; ++v) {
      int m = m0 + mo + v;
      if (m < M) {
        float x = acc[nf][v] + bv;
        if (act) x = silu_f(x);
        size_t o = (size_t)m * DIM + n;
        if (Obf) Obf[o] = (bf16)x;
        if (Of) Of[o] = x;
      }
    }
  }
}

// ---------------------------------------------------------------------------
// msg = H @ Wsi2 + bsi2 (3-wide GEMV); dsh = sh_scale(sph(dpos), msg); atomic acc.
__global__ void k_edge_msg0(const int* __restrict__ row, const int* __restrict__ col,
                            const float* __restrict__ pos, const bf16* __restrict__ H,
                            const float* __restrict__ W2, const float* __restrict__ b2,
                            float* __restrict__ sh_acc, int ebase, int ecount) {
  int el = blockIdx.x * 256 + threadIdx.x;
  if (el >= ecount) return;
  int e = ebase + el;
  int r = row[e], c = col[e];
  const bf16* h = H + (size_t)el * DIM;
  float m0 = b2[0], m1 = b2[1], m2 = b2[2];
  for (int k = 0; k < DIM; ++k) {
    float hv = (float)h[k];
    m0 += hv * W2[k * 3 + 0];
    m1 += hv * W2[k * 3 + 1];
    m2 += hv * W2[k * 3 + 2];
  }
  float s[9];
  sph9(pos[r * 3] - pos[c * 3], pos[r * 3 + 1] - pos[c * 3 + 1],
       pos[r * 3 + 2] - pos[c * 3 + 2], s);
  float w[9] = {m0, m1, m1, m1, m2, m2, m2, m2, m2};
  for (int i = 0; i < 9; ++i) atomicAdd(&sh_acc[(size_t)r * 9 + i], s[i] * w[i]);
}

__global__ void k_sh_mean(const float* __restrict__ acc, const float* __restrict__ cnt,
                          float* __restrict__ sh) {
  int i = blockIdx.x * 256 + threadIdx.x;
  if (i >= N_NODES * 9) return;
  sh[i] = acc[i] / fmaxf(cnt[i / 9], 1.f);
}

__global__ void k_node_center(const int* __restrict__ batch, const float* __restrict__ pos,
                              const bf16* __restrict__ H, const float* __restrict__ W2,
                              const float* __restrict__ b2, const float* __restrict__ gsum,
                              const float* __restrict__ gcnt, float* __restrict__ sh) {
  int n = blockIdx.x * 256 + threadIdx.x;
  if (n >= N_NODES) return;
  const bf16* h = H + (size_t)n * DIM;
  float m0 = b2[0], m1 = b2[1], m2 = b2[2];
  for (int k = 0; k < DIM; ++k) {
    float hv = (float)h[k];
    m0 += hv * W2[k * 3 + 0];
    m1 += hv * W2[k * 3 + 1];
    m2 += hv * W2[k * 3 + 2];
  }
  int g = batch[n];
  float gi = 1.f / fmaxf(gcnt[g], 1.f);
  float s[9];
  sph9(pos[n * 3] - gsum[g * 3] * gi, pos[n * 3 + 1] - gsum[g * 3 + 1] * gi,
       pos[n * 3 + 2] - gsum[g * 3 + 2] * gi, s);
  float w[9] = {m0, m1, m1, m1, m2, m2, m2, m2, m2};
  float* shn = sh + (size_t)n * 9;
  for (int i = 0; i < 9; ++i) shn[i] += s[i] * w[i];
  shn[0] = 0.f; shn[1] = 0.f; shn[2] = 0.f; shn[3] = 0.f;  // sh[:, :LMAX^2] = 0
}

// X1 row = [feat[row] | feat[col] | d2 | ip(3) | pad]
__global__ void k_build_x1(const int* __restrict__ row, const int* __restrict__ col,
                           const float* __restrict__ pos_cur, const float* __restrict__ sh,
                           const bf16* __restrict__ featb, bf16* __restrict__ X,
                           int ebase, int ecount) {
  long i = (long)blockIdx.x * 256 + threadIdx.x;
  if (i >= (long)ecount * K1PAD) return;
  int el = (int)(i / K1PAD), j = (int)(i % K1PAD);
  int e = ebase + el;
  int r = row[e], c = col[e];
  bf16 v = (bf16)0.f;
  if (j < DIM) {
    v = featb[(size_t)r * DIM + j];
  } else if (j < 2 * DIM) {
    v = featb[(size_t)c * DIM + (j - DIM)];
  } else if (j == 2 * DIM) {
    float dx = pos_cur[r * 3] - pos_cur[c * 3];
    float dy = pos_cur[r * 3 + 1] - pos_cur[c * 3 + 1];
    float dz = pos_cur[r * 3 + 2] - pos_cur[c * 3 + 2];
    v = (bf16)(dx * dx + dy * dy + dz * dz);
  } else if (j <= 2 * DIM + 3) {
    int ci = j - (2 * DIM + 1);
    const float* sr = sh + (size_t)r * 9;
    const float* sc = sh + (size_t)c * 9;
    float t = 0.f;
    if (ci == 0) t = sr[0] * sc[0];
    else if (ci == 1) { for (int q = 1; q < 4; ++q) t += sr[q] * sc[q]; }
    else { for (int q = 4; q < 9; ++q) t += sr[q] * sc[q]; }
    v = (bf16)t;
  }
  X[i] = v;
}

__global__ void k_m_agg(const int* __restrict__ row, const bf16* __restrict__ Mt,
                        float* __restrict__ m_agg, int ebase, int ecount) {
  long i = (long)blockIdx.x * 256 + threadIdx.x;
  if (i >= (long)ecount * DIM) return;
  int el = (int)(i >> 7), j = (int)(i & 127);
  int r = row[ebase + el];
  atomicAdd(&m_agg[(size_t)r * DIM + j], (float)Mt[i]);
}

__global__ void k_edge_p(const int* __restrict__ row, const int* __restrict__ col,
                         const float* __restrict__ pos_cur, const bf16* __restrict__ H,
                         const float* __restrict__ wp2, const float* __restrict__ bp2,
                         float* __restrict__ p_agg, int ebase, int ecount) {
  int el = blockIdx.x * 256 + threadIdx.x;
  if (el >= ecount) return;
  int e = ebase + el;
  const bf16* h = H + (size_t)el * DIM;
  float s = bp2[0];
  for (int k = 0; k < DIM; ++k) s += (float)h[k] * wp2[k];
  int r = row[e], c = col[e];
  atomicAdd(&p_agg[r * 3 + 0], (pos_cur[r * 3 + 0] - pos_cur[c * 3 + 0]) * s);
  atomicAdd(&p_agg[r * 3 + 1], (pos_cur[r * 3 + 1] - pos_cur[c * 3 + 1]) * s);
  atomicAdd(&p_agg[r * 3 + 2], (pos_cur[r * 3 + 2] - pos_cur[c * 3 + 2]) * s);
}

__global__ void k_edge_s(const int* __restrict__ row, const int* __restrict__ col,
                         const float* __restrict__ sh, const bf16* __restrict__ H,
                         const float* __restrict__ ws2, const float* __restrict__ bs2,
                         float* __restrict__ s_agg, int ebase, int ecount) {
  int el = blockIdx.x * 256 + threadIdx.x;
  if (el >= ecount) return;
  int e = ebase + el;
  const bf16* h = H + (size_t)el * DIM;
  float m0 = bs2[0], m1 = bs2[1], m2 = bs2[2];
  for (int k = 0; k < DIM; ++k) {
    float hv = (float)h[k];
    m0 += hv * ws2[k * 3 + 0];
    m1 += hv * ws2[k * 3 + 1];
    m2 += hv * ws2[k * 3 + 2];
  }
  int r = row[e], c = col[e];
  const float* sr = sh + (size_t)r * 9;
  const float* sc = sh + (size_t)c * 9;
  float w[9] = {m0, m1, m1, m1, m2, m2, m2, m2, m2};
  for (int i = 0; i < 9; ++i)
    atomicAdd(&s_agg[(size_t)r * 9 + i], (sr[i] - sc[i]) * w[i]);
}

__global__ void k_node_apply(const float* __restrict__ cnt, const float* __restrict__ p_agg,
                             const float* __restrict__ s_agg, float* __restrict__ pos_cur,
                             float* __restrict__ sh) {
  int n = blockIdx.x * 256 + threadIdx.x;
  if (n >= N_NODES) return;
  float inv = 1.f / fmaxf(cnt[n], 1.f);
  for (int i = 0; i < 3; ++i) pos_cur[n * 3 + i] += p_agg[n * 3 + i] * inv;
  for (int i = 0; i < 9; ++i) sh[n * 9 + i] += s_agg[n * 9 + i] * inv;
}

__global__ void k_build_x2(const float* __restrict__ feat, const float* __restrict__ m_agg,
                           const float* __restrict__ cnt, bf16* __restrict__ X2) {
  int i = blockIdx.x * 256 + threadIdx.x;
  if (i >= N_NODES * 256) return;
  int n = i >> 8, j = i & 255;
  bf16 v;
  if (j < DIM) v = (bf16)feat[(size_t)n * DIM + j];
  else v = (bf16)(m_agg[(size_t)n * DIM + (j - DIM)] / fmaxf(cnt[n], 1.f));
  X2[i] = v;
}

__global__ void k_out_init(float* __restrict__ out, const float* __restrict__ bpred) {
  int g = blockIdx.x * 256 + threadIdx.x;
  if (g < N_GRAPH) out[g] = bpred[0];
}

__global__ void k_pool_pred(const int* __restrict__ batch, const float* __restrict__ feat,
                            const float* __restrict__ pos_cur, const float* __restrict__ sh,
                            const float* __restrict__ Wpred, float* __restrict__ out) {
  int n = blockIdx.x * 256 + threadIdx.x;
  if (n >= N_NODES) return;
  float s = 0.f;
  const float* f = feat + (size_t)n * DIM;
  for (int j = 0; j < DIM; ++j) s += f[j] * Wpred[j];
  for (int i = 0; i < 3; ++i) s += pos_cur[n * 3 + i] * Wpred[DIM + i];
  for (int i = 0; i < 9; ++i) s += sh[n * 9 + i] * Wpred[DIM + 3 + i];
  atomicAdd(&out[batch[n]], s);
}

// ---------------------------------------------------------------------------
extern "C" void kernel_launch(void* const* d_in, const int* in_sizes, int n_in,
                              void* d_out, int out_size, void* d_ws, size_t ws_size,
                              hipStream_t stream) {
  (void)in_sizes; (void)n_in; (void)out_size; (void)ws_size;

  const int* atoms = (const int*)d_in[0];
  const int* row = (const int*)d_in[1];
  const int* col = row + N_EDGES;
  const int* batch = (const int*)d_in[2];
  const float* pos = (const float*)d_in[3];
  const float* emb = (const float*)d_in[4];
  const float* Wsi1 = (const float*)d_in[5];  const float* bsi1 = (const float*)d_in[6];
  const float* Wsi2 = (const float*)d_in[7];  const float* bsi2 = (const float*)d_in[8];
  const float* WsiC1 = (const float*)d_in[9]; const float* bsiC1 = (const float*)d_in[10];
  const float* WsiC2 = (const float*)d_in[11]; const float* bsiC2 = (const float*)d_in[12];
  const float* Wm1 = (const float*)d_in[13]; const float* bm1 = (const float*)d_in[14];
  const float* Wm2 = (const float*)d_in[15]; const float* bm2 = (const float*)d_in[16];
  const float* Wp1 = (const float*)d_in[17]; const float* bp1 = (const float*)d_in[18];
  const float* Wp2 = (const float*)d_in[19]; const float* bp2 = (const float*)d_in[20];
  const float* Wn1 = (const float*)d_in[21]; const float* bn1 = (const float*)d_in[22];
  const float* Wn2 = (const float*)d_in[23]; const float* bn2 = (const float*)d_in[24];
  const float* Ws1 = (const float*)d_in[25]; const float* bs1 = (const float*)d_in[26];
  const float* Ws2 = (const float*)d_in[27]; const float* bs2 = (const float*)d_in[28];
  const float* Wpred = (const float*)d_in[29]; const float* bpred = (const float*)d_in[30];
  float* out = (float*)d_out;

  // ---- workspace carve-out (256B aligned) ----
  char* p = (char*)d_ws;
  auto alloc = [&](size_t bytes) -> char* {
    char* r = p;
    p += (bytes + 255) & ~(size_t)255;
    return r;
  };
  float* feat = (float*)alloc((size_t)NPAD * DIM * 4);
  bf16* featb = (bf16*)alloc((size_t)NPAD * DIM * 2);
  float* sh = (float*)alloc((size_t)N_NODES * 9 * 4);
  float* pos_cur = (float*)alloc((size_t)N_NODES * 3 * 4);
  float* cnt = (float*)alloc((size_t)N_NODES * 4);
  float* gsum = (float*)alloc((size_t)N_GRAPH * 3 * 4);
  float* gcnt = (float*)alloc((size_t)N_GRAPH * 4);
  float* m_agg = (float*)alloc((size_t)N_NODES * DIM * 4);
  float* p_agg = (float*)alloc((size_t)N_NODES * 3 * 4);
  float* s_agg = (float*)alloc((size_t)N_NODES * 9 * 4);
  bf16* X1 = (bf16*)alloc((size_t)EC * K1PAD * 2);
  bf16* Hc = (bf16*)alloc((size_t)EC * DIM * 2);  // also reused as node hidden
  bf16* Mc = (bf16*)alloc((size_t)EC * DIM * 2);
  bf16* X2 = (bf16*)alloc((size_t)NPAD * 256 * 2);
  bf16* wsi1t = (bf16*)alloc((size_t)DIM * K1PAD * 2);
  bf16* wsic1t = (bf16*)alloc((size_t)DIM * DIM * 2);
  bf16* wl[NLAYERS][6];
  for (int l = 0; l < NLAYERS; ++l) {
    wl[l][0] = (bf16*)alloc((size_t)DIM * K1PAD * 2);  // Wm1^T (K=288)
    wl[l][1] = (bf16*)alloc((size_t)DIM * DIM * 2);    // Wm2^T
    wl[l][2] = (bf16*)alloc((size_t)DIM * DIM * 2);    // Wp1^T
    wl[l][3] = (bf16*)alloc((size_t)DIM * DIM * 2);    // Ws1^T
    wl[l][4] = (bf16*)alloc((size_t)DIM * 256 * 2);    // Wn1^T (K=256)
    wl[l][5] = (bf16*)alloc((size_t)DIM * DIM * 2);    // Wn2^T
  }

  auto g1 = [](long n) { return dim3((unsigned)((n + 255) / 256)); };
#define ZERO(ptr, n) k_zero<<<g1((long)(n)), 256, 0, stream>>>((float*)(ptr), (long)(n))

  // ---- pack weights to transposed/padded bf16 ----
  k_pack_wt<<<g1(DIM * K1PAD), 256, 0, stream>>>(Wsi1, wsi1t, 257, K1PAD);
  k_pack_wt<<<g1(DIM * DIM), 256, 0, stream>>>(WsiC1, wsic1t, DIM, DIM);
  for (int l = 0; l < NLAYERS; ++l) {
    k_pack_wt<<<g1(DIM * K1PAD), 256, 0, stream>>>(Wm1 + (size_t)l * 260 * DIM, wl[l][0], 260, K1PAD);
    k_pack_wt<<<g1(DIM * DIM), 256, 0, stream>>>(Wm2 + (size_t)l * DIM * DIM, wl[l][1], DIM, DIM);
    k_pack_wt<<<g1(DIM * DIM), 256, 0, stream>>>(Wp1 + (size_t)l * DIM * DIM, wl[l][2], DIM, DIM);
    k_pack_wt<<<g1(DIM * DIM), 256, 0, stream>>>(Ws1 + (size_t)l * DIM * DIM, wl[l][3], DIM, DIM);
    k_pack_wt<<<g1(DIM * 256), 256, 0, stream>>>(Wn1 + (size_t)l * 256 * DIM, wl[l][4], 256, 256);
    k_pack_wt<<<g1(DIM * DIM), 256, 0, stream>>>(Wn2 + (size_t)l * DIM * DIM, wl[l][5], DIM, DIM);
  }

  // ---- init ----
  ZERO(cnt, N_NODES);
  ZERO(gsum, N_GRAPH * 3);
  ZERO(gcnt, N_GRAPH);
  ZERO(s_agg, N_NODES * 9);  // reused as initial sh accumulator
  k_embed<<<g1((long)N_NODES * DIM), 256, 0, stream>>>(atoms, batch, emb, pos, feat,
                                                       featb, pos_cur, gsum, gcnt);
  k_edge_count<<<g1(N_EDGES), 256, 0, stream>>>(row, cnt);

  // ---- initial edge messages -> sh ----
  for (int c0 = 0; c0 < NCHUNKS; ++c0) {
    int eb = c0 * EC;
    k_build_x0<<<g1((long)EC * K1PAD), 256, 0, stream>>>(row, col, pos, featb, X1, eb, EC);
    k_gemm_n128<<<dim3(EC / 128), 256, 0, stream>>>(X1, K1PAD, EC, wsi1t, K1PAD, bsi1,
                                                    Hc, nullptr, 1);
    k_edge_msg0<<<g1(EC), 256, 0, stream>>>(row, col, pos, Hc, Wsi2, bsi2, s_agg, eb, EC);
  }
  k_sh_mean<<<g1(N_NODES * 9), 256, 0, stream>>>(s_agg, cnt, sh);

  // ---- center-of-mass SH term ----
  k_gemm_n128<<<dim3((N_NODES + 127) / 128), 256, 0, stream>>>(featb, DIM, N_NODES, wsic1t,
                                                               DIM, bsiC1, Hc, nullptr, 1);
  k_node_center<<<g1(N_NODES), 256, 0, stream>>>(batch, pos, Hc, WsiC2, bsiC2, gsum, gcnt, sh);

  // ---- message-passing layers ----
  for (int l = 0; l < NLAYERS; ++l) {
    ZERO(m_agg, (long)N_NODES * DIM);
    ZERO(p_agg, N_NODES * 3);
    ZERO(s_agg, N_NODES * 9);
    const float* bm1l = bm1 + (size_t)l * DIM;
    const float* bm2l = bm2 + (size_t)l * DIM;
    const float* bp1l = bp1 + (size_t)l * DIM;
    const float* bp2l = bp2 + l;
    const float* bs1l = bs1 + (size_t)l * DIM;
    const float* bs2l = bs2 + (size_t)l * 3;
    const float* bn1l = bn1 + (size_t)l * DIM;
    const float* bn2l = bn2 + (size_t)l * DIM;
    const float* wp2l = Wp2 + (size_t)l * DIM;
    const float* ws2l = Ws2 + (size_t)l * DIM * 3;

    for (int c0 = 0; c0 < NCHUNKS; ++c0) {
      int eb = c0 * EC;
      k_build_x1<<<g1((long)EC * K1PAD), 256, 0, stream>>>(row, col, pos_cur, sh, featb,
                                                           X1, eb, EC);
      k_gemm_n128<<<dim3(EC / 128), 256, 0, stream>>>(X1, K1PAD, EC, wl[l][0], K1PAD,
                                                      bm1l, Hc, nullptr, 1);
      k_gemm_n128<<<dim3(EC / 128), 256, 0, stream>>>(Hc, DIM, EC, wl[l][1], DIM,
                                                      bm2l, Mc, nullptr, 1);
      k_m_agg<<<g1((long)EC * DIM), 256, 0, stream>>>(row, Mc, m_agg, eb, EC);
      k_gemm_n128<<<dim3(EC / 128), 256, 0, stream>>>(Mc, DIM, EC, wl[l][2], DIM,
                                                      bp1l, Hc, nullptr, 1);
      k_edge_p<<<g1(EC), 256, 0, stream>>>(row, col, pos_cur, Hc, wp2l, bp2l, p_agg, eb, EC);
      k_gemm_n128<<<dim3(EC / 128), 256, 0, stream>>>(Mc, DIM, EC, wl[l][3], DIM,
                                                      bs1l, Hc, nullptr, 1);
      k_edge_s<<<g1(EC), 256, 0, stream>>>(row, col, sh, Hc, ws2l, bs2l, s_agg, eb, EC);
    }
    k_node_apply<<<g1(N_NODES), 256, 0, stream>>>(cnt, p_agg, s_agg, pos_cur, sh);
    k_build_x2<<<g1((long)N_NODES * 256), 256, 0, stream>>>(feat, m_agg, cnt, X2);
    k_gemm_n128<<<dim3((N_NODES + 127) / 128), 256, 0, stream>>>(X2, 256, N_NODES, wl[l][4],
                                                                 256, bn1l, Hc, nullptr, 1);
    k_gemm_n128<<<dim3((N_NODES + 127) / 128), 256, 0, stream>>>(Hc, DIM, N_NODES, wl[l][5],
                                                                 DIM, bn2l, featb, feat, 0);
  }

  // ---- pooling + prediction ----
  k_out_init<<<1, 256, 0, stream>>>(out, bpred);
  k_pool_pred<<<g1(N_NODES), 256, 0, stream>>>(batch, feat, pos_cur, sh, Wpred, out);
#undef ZERO
}
